// LstmNet2_62319975465210
// MI455X (gfx1250) — matrix operand order repaired and checked by reading, not compiled, not run on hardware
//
#include <hip/hip_runtime.h>

#define HIDDEN 10
#define NGATE  40
#define NCLS   10

typedef __attribute__((ext_vector_type(16))) _Float16 v16h;
typedef __attribute__((ext_vector_type(8)))  _Float16 v8h;
typedef __attribute__((ext_vector_type(8)))  float    v8f;

__device__ __forceinline__ float rcpf(float x) { return __builtin_amdgcn_rcpf(x); }
__device__ __forceinline__ float sigf(float x) { return rcpf(1.0f + __expf(-x)); }
__device__ __forceinline__ float tanhfast(float x) {
    // 1 - 2/(e^{2x}+1): saturates correctly to +-1 for large |x|
    float e = __expf(2.0f * x);
    return fmaf(-2.0f, rcpf(e + 1.0f), 1.0f);
}
__device__ __forceinline__ v8f wmma_f16(v16h a, v16h b, v8f c) {
    return __builtin_amdgcn_wmma_f32_16x16x32_f16(false, a, false, b,
                                                  (short)0, c, false, false);
}

// One wave (32 lanes) per block; each wave owns 16 batch rows and runs the
// whole fused 3-layer LSTM + MLP head over the T-step recurrence.
// Per layer, per step: ONE wmma per 16-wide N tile over the K-concatenated
// operand [x|h_in (K=0..9), h_prev (K=10..19), 0 pad] x [Wih; Whh].
__global__ __launch_bounds__(32) void lstm3_fused(
    const float* __restrict__ x,
    const float* __restrict__ Wih0, const float* __restrict__ Whh0,
    const float* __restrict__ bih0, const float* __restrict__ bhh0,
    const float* __restrict__ Wih1, const float* __restrict__ Whh1,
    const float* __restrict__ bih1, const float* __restrict__ bhh1,
    const float* __restrict__ Wih2, const float* __restrict__ Whh2,
    const float* __restrict__ bih2, const float* __restrict__ bhh2,
    const float* __restrict__ W1,   const float* __restrict__ b1,
    const float* __restrict__ W2,   const float* __restrict__ b2,
    float* __restrict__ out, int T)
{
    // K-concatenated A staging rows: [0..9]=input (x or h_{l-1}(t)),
    // [10..19]=h_l(t-1), [20..31]=0 forever (WMMA K padding).
    __shared__ __align__(16) _Float16 comb[3][16][32];
    __shared__ __align__(16) float    gbufP[16][HIDDEN][4]; // [row][j][i,f,g,o]
    __shared__ __align__(16) float    hf2[16][HIDDEN];      // f32 h2 at t = T-1
    __shared__ __align__(16) float    z1s[16][HIDDEN];

    const int lane  = threadIdx.x;
    const int n_loc = lane & 15;   // N column (B/C/D) or M row (A)
    const int hi    = lane >> 4;
    const int mbase = hi * 8;      // D rows this lane holds: mbase..mbase+7
    const int r0    = blockIdx.x * 16;

    for (int i = lane; i < 3 * 16 * 32; i += 32)
        ((_Float16*)comb)[i] = (_Float16)0;

    const float* WihA[3] = { Wih0, Wih1, Wih2 };
    const float* WhhA[3] = { Whh0, Whh1, Whh2 };
    const float* bihA[3] = { bih0, bih1, bih2 };
    const float* bhhA[3] = { bhh0, bhh1, bhh2 };
    const int    dinA[3] = { 1, HIDDEN, HIDDEN };

    // Loop-invariant C operands: bias broadcast in D layout (N = n_loc).
    v8f cbias[3][3];
    // B fragments: 32x16 f16 B in this file's contiguous K order; lanes 0-15 hold K=0..15,
    // lanes 16-31 hold K=16..31.  K0..9 = Wih row, K10..19 = Whh row.
    v16h Bf[3][3];
#pragma unroll
    for (int l = 0; l < 3; ++l) {
#pragma unroll
        for (int tt = 0; tt < 3; ++tt) {
            int n = tt * 16 + n_loc;
            bool ok = n < NGATE;
            float bv = ok ? (bihA[l][n] + bhhA[l][n]) : 0.0f;
#pragma unroll
            for (int j = 0; j < 8; ++j) cbias[l][tt][j] = bv;

            v16h b;
#pragma unroll
            for (int k = 0; k < 16; ++k) b[k] = (_Float16)0;
            if (ok) {
                if (hi == 0) {
                    for (int k = 0; k < dinA[l]; ++k)
                        b[k] = (_Float16)WihA[l][n * dinA[l] + k];     // K=0..din-1
#pragma unroll
                    for (int k = 0; k < 6; ++k)
                        b[10 + k] = (_Float16)WhhA[l][n * HIDDEN + k]; // K=10..15
                } else {
#pragma unroll
                    for (int k = 0; k < 4; ++k)
                        b[k] = (_Float16)WhhA[l][n * HIDDEN + 6 + k];  // K=16..19
                }
            }
            Bf[l][tt] = b;
        }
    }

    // cell state: 160 (row, j) elements spread 5 per lane
    float cr[3][5];
#pragma unroll
    for (int l = 0; l < 3; ++l)
#pragma unroll
        for (int s = 0; s < 5; ++s) cr[l][s] = 0.0f;

    // A fragment: lane<16 -> K0..15, lane>=16 -> K16..31
    // (two ds b128 reads per lane)
    auto mk_a = [&](int l) {
        v8h lo = *(const v8h*)(&comb[l][n_loc][2 * mbase]);
        v8h hh = *(const v8h*)(&comb[l][n_loc][2 * mbase + 8]);
        v16h a;
#pragma unroll
        for (int k = 0; k < 8; ++k) { a[k] = lo[k]; a[k + 8] = hh[k]; }
        return a;
    };

    // Scatter D tiles into gate-packed layout; per-lane gate index and its
    // (j, kind) decomposition are loop-invariant (hoisted out of the T loop).
    auto store_gates = [&](const v8f* d) {
#pragma unroll
        for (int tt = 0; tt < 3; ++tt) {
            int g  = tt * 16 + n_loc;
            int gk = g / HIDDEN;            // gate kind: 0=i 1=f 2=g 3=o
            int gj = g - gk * HIDDEN;       // hidden index
            if (g < NGATE) {
#pragma unroll
                for (int j = 0; j < 8; ++j)
                    gbufP[mbase + j][gj][gk] = d[tt][j];
            }
        }
    };

    auto cell = [&](int l, float* c_, bool save_last) {
#pragma unroll
        for (int s = 0; s < 5; ++s) {
            int e   = s * 32 + lane;            // 0..159
            int row = e / HIDDEN;
            int j   = e - row * HIDDEN;
            float4 gv = *(const float4*)(&gbufP[row][j][0]);  // one b128
            float cc = sigf(gv.y) * c_[s] + sigf(gv.x) * tanhfast(gv.z);
            c_[s] = cc;
            float hh = sigf(gv.w) * tanhfast(cc);
            comb[l][row][10 + j] = (_Float16)hh;           // own recurrent slot
            if (l < 2) comb[l + 1][row][j] = (_Float16)hh; // next layer input
            if (save_last) hf2[row][j] = hh;
        }
    };

    __syncthreads();

    for (int t = 0; t < T; ++t) {
        // scalar input rides the WMMA at K=0 of layer 0
        if (lane < 16) {
            const float* xp = &x[(size_t)(r0 + lane) * (size_t)T + t];
            float xv = *xp;
            if (t + 64 < T) __builtin_prefetch(xp + 64, 0, 0);
            comb[0][lane][0] = (_Float16)xv;
        }
        __syncthreads();

#pragma unroll
        for (int l = 0; l < 3; ++l) {
            v16h a = mk_a(l);
            v8f d[3];
#pragma unroll
            for (int tt = 0; tt < 3; ++tt)
                d[tt] = wmma_f16(a, Bf[l][tt], cbias[l][tt]);
            __syncthreads();          // gbufP consumers of previous layer done
            store_gates(d);
            __syncthreads();
            cell(l, cr[l], (l == 2) && (t == T - 1));
            __syncthreads();
        }
    }

    // ---------- MLP head on last-step h2 (f32 path)
#pragma unroll
    for (int s = 0; s < 5; ++s) {
        int e = s * 32 + lane; int row = e / HIDDEN; int k = e - row * HIDDEN;
        float acc = b1[k];
#pragma unroll
        for (int j = 0; j < HIDDEN; ++j)
            acc = fmaf(hf2[row][j], W1[k * HIDDEN + j], acc);
        z1s[row][k] = fmaxf(acc, 0.0f);
    }
    __syncthreads();
#pragma unroll
    for (int s = 0; s < 5; ++s) {
        int e = s * 32 + lane; int row = e / HIDDEN; int n = e - row * HIDDEN;
        float acc = b2[n];
#pragma unroll
        for (int k = 0; k < HIDDEN; ++k)
            acc = fmaf(z1s[row][k], W2[n * HIDDEN + k], acc);
        out[(size_t)(r0 + row) * NCLS + n] = acc;
    }
}

extern "C" void kernel_launch(void* const* d_in, const int* in_sizes, int n_in,
                              void* d_out, int out_size, void* d_ws, size_t ws_size,
                              hipStream_t stream) {
    const float* x    = (const float*)d_in[0];
    const float* Wih0 = (const float*)d_in[1];
    const float* Whh0 = (const float*)d_in[2];
    const float* bih0 = (const float*)d_in[3];
    const float* bhh0 = (const float*)d_in[4];
    const float* Wih1 = (const float*)d_in[5];
    const float* Whh1 = (const float*)d_in[6];
    const float* bih1 = (const float*)d_in[7];
    const float* bhh1 = (const float*)d_in[8];
    const float* Wih2 = (const float*)d_in[9];
    const float* Whh2 = (const float*)d_in[10];
    const float* bih2 = (const float*)d_in[11];
    const float* bhh2 = (const float*)d_in[12];
    const float* W1   = (const float*)d_in[13];
    const float* b1   = (const float*)d_in[14];
    const float* W2   = (const float*)d_in[15];
    const float* b2   = (const float*)d_in[16];

    const int B = out_size / NCLS;        // 2048
    const int T = in_sizes[0] / B;        // 2048

    dim3 grid(B / 16), block(32);
    lstm3_fused<<<grid, block, 0, stream>>>(
        x, Wih0, Whh0, bih0, bhh0, Wih1, Whh1, bih1, bhh1,
        Wih2, Whh2, bih2, bhh2, W1, b1, W2, b2, (float*)d_out, T);
}
